// NAMTuringOnlyJump_5076651344119
// MI455X (gfx1250) — matrix operand order repaired and checked by reading, not compiled, not run on hardware
//
#include <hip/hip_runtime.h>
#include <cstdint>

typedef __attribute__((ext_vector_type(16))) __bf16 v16bf;
typedef __attribute__((ext_vector_type(8)))  __bf16 v8bf;
typedef __attribute__((ext_vector_type(8)))  float  v8f;

#define S_  1024
#define B_  32
#define D_  512
#define NT_ 4
#define HD_ 64
#define TAPE_ 64
#define SB_ (S_ * B_)

__device__ __forceinline__ float sigf(float x) { return 1.0f / (1.0f + expf(-x)); }

// ---------------------------------------------------------------------------
// f32 -> bf16 conversion
__global__ void cvt_f32_bf16(const float* __restrict__ src, __bf16* __restrict__ dst, size_t n) {
    size_t i = (size_t)blockIdx.x * blockDim.x + threadIdx.x;
    if (i < n) dst[i] = (__bf16)src[i];
}

// f32 -> bf16 with row padding (used to pad W_act 24 rows -> 64 rows)
__global__ void cvt_pad_rows(const float* __restrict__ src, __bf16* __restrict__ dst,
                             int rows_src, int rows_dst, int cols) {
    int i = blockIdx.x * blockDim.x + threadIdx.x;
    if (i >= rows_dst * cols) return;
    int row = i / cols;
    dst[i] = (row < rows_src) ? (__bf16)src[i] : (__bf16)0.0f;
}

__global__ void add_vec(const float* __restrict__ a, const float* __restrict__ b,
                        float* __restrict__ o, int n) {
    int i = blockIdx.x * blockDim.x + threadIdx.x;
    if (i < n) o[i] = a[i] + b[i];
}

__global__ void fill_bf16_zero(__bf16* p, size_t n) {
    size_t i = (size_t)blockIdx.x * blockDim.x + threadIdx.x;
    if (i < n) p[i] = (__bf16)0.0f;
}
__global__ void fill_f32_zero(float* p, size_t n) {
    size_t i = (size_t)blockIdx.x * blockDim.x + threadIdx.x;
    if (i < n) p[i] = 0.0f;
}

// ---------------------------------------------------------------------------
// C[M x N] = A[M x K](bf16,row-major) * B[N x K](bf16,row-major == W, used as W^T) + bias
// One wave per 16x64 C strip (4 N-tiles): A fragment loaded once per k-step and
// reused across 4 v_wmma_f32_16x16x32_bf16. N must be a multiple of 64.
// VGPR layouts per CDNA5 ISA 7.12.2.
__global__ void gemm_bf16_wmma(const __bf16* __restrict__ A, int lda,
                               const __bf16* __restrict__ Bm, int ldb,
                               float* __restrict__ C, int ldc,
                               const float* __restrict__ bias, int nbias,
                               int M, int N, int K) {
    int wave = (int)((blockIdx.x * blockDim.x + threadIdx.x) >> 5);
    int lane = threadIdx.x & 31;
    int nquads = N >> 6;                 // strips of 4 tiles
    int total = (M >> 4) * nquads;
    if (wave >= total) return;
    int mtile = wave / nquads;
    int nquad = wave % nquads;
    int r  = lane & 15;
    int hi = lane >> 4;

    const __bf16* arow = A + (size_t)(mtile * 16 + r) * lda;
    const __bf16* brow0 = Bm + (size_t)(nquad * 64 + r) * ldb;

    v8f acc[4] = {};
    for (int k0 = 0; k0 < K; k0 += 32) {
        // A: lanes<16 -> K {0..7, 16..23}; lanes>=16 -> K {8..15, 24..31}
        v8bf a0 = *(const v8bf*)(arow + k0 + hi * 8);
        v8bf a1 = *(const v8bf*)(arow + k0 + 16 + hi * 8);
        v16bf a = __builtin_shufflevector(a0, a1, 0,1,2,3,4,5,6,7,8,9,10,11,12,13,14,15);
#pragma unroll
        for (int q = 0; q < 4; ++q) {
            // B (=W^T): lanes<16 -> K {0..7, 16..23}; lanes>=16 -> K {8..15, 24..31} from W row n
            const __bf16* brow = brow0 + (size_t)(q * 16) * ldb;
            v8bf b0 = *(const v8bf*)(brow + k0 + hi * 8);
            v8bf b1 = *(const v8bf*)(brow + k0 + 16 + hi * 8);
            v16bf b = __builtin_shufflevector(b0, b1, 0,1,2,3,4,5,6,7,8,9,10,11,12,13,14,15);
            acc[q] = __builtin_amdgcn_wmma_f32_16x16x32_bf16(false, a, false, b, (short)0, acc[q], false, false);
        }
    }
#pragma unroll
    for (int q = 0; q < 4; ++q) {
        int col = nquad * 64 + q * 16 + r;
        float bv = (bias && col < nbias) ? bias[col] : 0.0f;
#pragma unroll
        for (int i = 0; i < 8; ++i) {
            int row = mtile * 16 + hi * 8 + i;   // C VGPR i: M=i (lanes<16), M=i+8 (lanes>=16)
            C[(size_t)row * ldc + col] = acc[q][i] + bv;
        }
    }
}

// ---------------------------------------------------------------------------
// One LSTM step: g = gx_s + h_prev @ W_hh^T ; gates -> (c,h).
// 64 waves total: wave = mtile(2) x dtile(32). Each wave does 4 WMMA accums
// (one per gate) per k-step so the nonlinearity has all 4 gates locally.
__global__ void lstm_step(const float* __restrict__ gx_s,     // (32 x 2048), biases included
                          const __bf16* __restrict__ hprev,   // (32 x 512)
                          const __bf16* __restrict__ Whh,     // (2048 x 512)
                          float* __restrict__ c_state,        // (32 x 512)
                          __bf16* __restrict__ hout) {        // (32 x 512) = hidden_bf[s]
    int wave = (int)((blockIdx.x * blockDim.x + threadIdx.x) >> 5); // 0..63
    int lane = threadIdx.x & 31;
    int mtile = wave >> 5;   // 0..1
    int dt    = wave & 31;   // 0..31 (d tile over 512)
    int r  = lane & 15;
    int hi = lane >> 4;

    const __bf16* arow = hprev + (size_t)(mtile * 16 + r) * 512;
    v8f acc[4] = {};
    for (int k0 = 0; k0 < 512; k0 += 32) {
        v8bf a0 = *(const v8bf*)(arow + k0 + hi * 8);
        v8bf a1 = *(const v8bf*)(arow + k0 + 16 + hi * 8);
        v16bf a = __builtin_shufflevector(a0, a1, 0,1,2,3,4,5,6,7,8,9,10,11,12,13,14,15);
#pragma unroll
        for (int g = 0; g < 4; ++g) {
            const __bf16* brow = Whh + (size_t)(g * 512 + dt * 16 + r) * 512;
            v8bf b0 = *(const v8bf*)(brow + k0 + hi * 8);
            v8bf b1 = *(const v8bf*)(brow + k0 + 16 + hi * 8);
            v16bf b = __builtin_shufflevector(b0, b1, 0,1,2,3,4,5,6,7,8,9,10,11,12,13,14,15);
            acc[g] = __builtin_amdgcn_wmma_f32_16x16x32_bf16(false, a, false, b, (short)0, acc[g], false, false);
        }
    }
    int d = dt * 16 + r;
#pragma unroll
    for (int i = 0; i < 8; ++i) {
        int m = mtile * 16 + hi * 8 + i;
        const float* gx = gx_s + (size_t)m * 2048;
        float gi = acc[0][i] + gx[d];
        float gf = acc[1][i] + gx[512 + d];
        float gg = acc[2][i] + gx[1024 + d];
        float go = acc[3][i] + gx[1536 + d];
        size_t idx = (size_t)m * 512 + d;
        float c = sigf(gf) * c_state[idx] + sigf(gi) * tanhf(gg);
        c_state[idx] = c;
        hout[idx] = (__bf16)(sigf(go) * tanhf(c));
    }
}

// ---------------------------------------------------------------------------
// Normalize each contiguous group of 64 floats to unit L2 norm (keys/queries).
__global__ void rownorm64(float* __restrict__ buf, int nrows) {
    int row = blockIdx.x * blockDim.x + threadIdx.x;
    if (row >= nrows) return;
    float* p = buf + (size_t)row * 64;
    float s = 0.0f;
#pragma unroll 8
    for (int j = 0; j < 64; ++j) s += p[j] * p[j];
    float inv = 1.0f / fmaxf(sqrtf(s), 1e-12f);
#pragma unroll 8
    for (int j = 0; j < 64; ++j) p[j] *= inv;
}

// actions (raw, 24-wide padded to 64) -> per-(s,b,nt) [rd0,rd1,wd0,wd1,rw0,wprob]
__global__ void postact(const float* __restrict__ ract, float* __restrict__ acts, int n) {
    int i = blockIdx.x * blockDim.x + threadIdx.x;   // over S*B*NT
    if (i >= n) return;
    int nt = i & 3;
    size_t sb = (size_t)(i >> 2);
    const float* a = ract + sb * 64 + nt * 6;
    float* o = acts + (size_t)i * 6;
    float m0 = fmaxf(a[0], a[1]);
    float e0 = expf(a[0] - m0), e1 = expf(a[1] - m0);
    o[0] = e0 / (e0 + e1); o[1] = e1 / (e0 + e1);
    float m1 = fmaxf(a[2], a[3]);
    float f0 = expf(a[2] - m1), f1 = expf(a[3] - m1);
    o[2] = f0 / (f0 + f1); o[3] = f1 / (f0 + f1);
    o[4] = sigf(a[4]);
    o[5] = sigf(a[5]);
}

// ---------------------------------------------------------------------------
// Turing tape scan: one workgroup per (b,nt); 64x64 tape + key-tape in LDS.
// Thread t owns tape column c=t (all column updates are race-free); barriers
// only around the row-direction jpos reduction.
__global__ void turing_scan(const float* __restrict__ values,
                            const float* __restrict__ keys,
                            const float* __restrict__ queries,
                            const float* __restrict__ acts,
                            float* __restrict__ read_outs,
                            float* __restrict__ tape_out) {
    int b  = blockIdx.x >> 2;
    int nt = blockIdx.x & 3;
    int t  = threadIdx.x;            // 0..63
    __shared__ float tape[TAPE_ * 64];
    __shared__ float tkey[TAPE_ * 64];
    __shared__ float rpos[TAPE_], wpos[TAPE_], qbuf[64], jbuf[TAPE_];

    for (int l = 0; l < TAPE_; ++l) { tape[l * 64 + t] = 0.0f; tkey[l * 64 + t] = 0.0f; }
    rpos[t] = (t == 0) ? 1.0f : 0.0f;
    wpos[t] = rpos[t];
    __syncthreads();

    for (int s = 0; s < S_; ++s) {
        size_t sb = (size_t)s * B_ + b;
        const float* ac = acts + (sb * NT_ + nt) * 6;
        float rd0 = ac[0], rd1 = ac[1], wd0 = ac[2], wd1 = ac[3], rw0 = ac[4], wprob = ac[5];
        size_t base = sb * (NT_ * HD_) + nt * HD_ + t;
        float v  = values[base];
        float kk = keys[base];
        qbuf[t] = queries[base];
        __syncthreads();

        // old value / old key on column t (pre-update tape)
        float ov = 0.0f, ok = 0.0f;
        for (int l = 0; l < TAPE_; ++l) {
            ov += tape[l * 64 + t] * wpos[l];
            ok += tkey[l * 64 + t] * wpos[l];
        }
        float dv = (v - ov) * wprob;
        float dk = (kk - ok) * wprob;
        // update column t, read out with updated tape
        float rt = 0.0f;
        for (int l = 0; l < TAPE_; ++l) {
            float tv = tape[l * 64 + t] + wpos[l] * dv;
            tape[l * 64 + t] = tv;
            rt += tv * rpos[l];
            tkey[l * 64 + t] += wpos[l] * dk;
        }
        read_outs[base] = rt * rw0;
        __syncthreads();

        // jpos over row t (needs fully updated tkey)
        float jp = 0.0f;
        for (int c = 0; c < 64; ++c) jp += tkey[t * 64 + c] * qbuf[c];
        jbuf[t] = jp;
        __syncthreads();
        float jn2 = 0.0f;
        for (int l = 0; l < TAPE_; ++l) jn2 += jbuf[l] * jbuf[l];
        float jpn = jp / fmaxf(sqrtf(jn2), 1e-12f);
        float wn = wpos[t] * wd0 + jpn * wd1;
        float rn = rpos[t] * rd0 + jpn * rd1;
        __syncthreads();
        wpos[t] = wn;
        rpos[t] = rn;
        __syncthreads();
    }
    // tape reshaped (tapelen, B, NT*HD): idx = l*B*256 + b*256 + nt*64 + t
    for (int l = 0; l < TAPE_; ++l)
        tape_out[(size_t)l * B_ * (NT_ * HD_) + (size_t)b * (NT_ * HD_) + nt * HD_ + t] =
            tape[l * 64 + t];
}

// ---------------------------------------------------------------------------
extern "C" void kernel_launch(void* const* d_in, const int* in_sizes, int n_in,
                              void* d_out, int out_size, void* d_ws, size_t ws_size,
                              hipStream_t stream) {
    const float* inputs = (const float*)d_in[0];   // (S,B,D)
    const float* W_ih   = (const float*)d_in[1];   // (2048,512)
    const float* W_hh   = (const float*)d_in[2];   // (2048,512)
    const float* b_ih   = (const float*)d_in[3];
    const float* b_hh   = (const float*)d_in[4];
    const float* W_act  = (const float*)d_in[5];   // (24,512)
    const float* b_act  = (const float*)d_in[6];
    const float* W_val  = (const float*)d_in[7];   // (256,512)
    const float* b_val  = (const float*)d_in[8];
    const float* W_key  = (const float*)d_in[9];
    const float* b_key  = (const float*)d_in[10];
    const float* W_q    = (const float*)d_in[11];
    const float* b_q    = (const float*)d_in[12];
    const float* W_out  = (const float*)d_in[13];  // (512,256)
    const float* b_out  = (const float*)d_in[14];
    (void)in_sizes; (void)n_in; (void)out_size; (void)ws_size;

    // workspace carve-out (256B aligned)
    uint8_t* ws = (uint8_t*)d_ws;
    size_t off = 0;
    auto alloc = [&](size_t bytes) { size_t o = off; off += (bytes + 255) & ~(size_t)255; return o; };
    __bf16* Xbf   = (__bf16*)(ws + alloc((size_t)SB_ * D_ * 2));
    __bf16* Hbf   = (__bf16*)(ws + alloc((size_t)SB_ * D_ * 2));
    float*  GX    = (float*) (ws + alloc((size_t)SB_ * 2048 * 4));
    float*  VAL   = (float*) (ws + alloc((size_t)SB_ * 256 * 4));
    float*  KEY   = (float*) (ws + alloc((size_t)SB_ * 256 * 4));
    float*  QRY   = (float*) (ws + alloc((size_t)SB_ * 256 * 4));
    float*  RACT  = (float*) (ws + alloc((size_t)SB_ * 64 * 4));
    float*  ACTS  = (float*) (ws + alloc((size_t)SB_ * NT_ * 6 * 4));
    float*  ROUT  = (float*) (ws + alloc((size_t)SB_ * 256 * 4));
    __bf16* ROUTB = (__bf16*)(ws + alloc((size_t)SB_ * 256 * 2));
    __bf16* WIHb  = (__bf16*)(ws + alloc((size_t)2048 * 512 * 2));
    __bf16* WHHb  = (__bf16*)(ws + alloc((size_t)2048 * 512 * 2));
    __bf16* WVALb = (__bf16*)(ws + alloc((size_t)256 * 512 * 2));
    __bf16* WKEYb = (__bf16*)(ws + alloc((size_t)256 * 512 * 2));
    __bf16* WQb   = (__bf16*)(ws + alloc((size_t)256 * 512 * 2));
    __bf16* WACTb = (__bf16*)(ws + alloc((size_t)64 * 512 * 2));
    __bf16* WOUTb = (__bf16*)(ws + alloc((size_t)512 * 256 * 2));
    float*  BIASG = (float*) (ws + alloc((size_t)2048 * 4));
    __bf16* HZERO = (__bf16*)(ws + alloc((size_t)B_ * 512 * 2));
    float*  CST   = (float*) (ws + alloc((size_t)B_ * 512 * 4));

    auto cdiv = [](size_t a, size_t b) { return (unsigned)((a + b - 1) / b); };

    // 1) conversions
    cvt_f32_bf16<<<cdiv((size_t)SB_ * D_, 256), 256, 0, stream>>>(inputs, Xbf, (size_t)SB_ * D_);
    cvt_f32_bf16<<<cdiv((size_t)2048 * 512, 256), 256, 0, stream>>>(W_ih, WIHb, (size_t)2048 * 512);
    cvt_f32_bf16<<<cdiv((size_t)2048 * 512, 256), 256, 0, stream>>>(W_hh, WHHb, (size_t)2048 * 512);
    cvt_f32_bf16<<<cdiv((size_t)256 * 512, 256), 256, 0, stream>>>(W_val, WVALb, (size_t)256 * 512);
    cvt_f32_bf16<<<cdiv((size_t)256 * 512, 256), 256, 0, stream>>>(W_key, WKEYb, (size_t)256 * 512);
    cvt_f32_bf16<<<cdiv((size_t)256 * 512, 256), 256, 0, stream>>>(W_q, WQb, (size_t)256 * 512);
    cvt_f32_bf16<<<cdiv((size_t)512 * 256, 256), 256, 0, stream>>>(W_out, WOUTb, (size_t)512 * 256);
    cvt_pad_rows<<<cdiv((size_t)64 * 512, 256), 256, 0, stream>>>(W_act, WACTb, 24, 64, 512);
    add_vec<<<cdiv(2048, 256), 256, 0, stream>>>(b_ih, b_hh, BIASG, 2048);
    fill_bf16_zero<<<cdiv((size_t)B_ * 512, 256), 256, 0, stream>>>(HZERO, (size_t)B_ * 512);
    fill_f32_zero<<<cdiv((size_t)B_ * 512, 256), 256, 0, stream>>>(CST, (size_t)B_ * 512);

    // 2) input projections (WMMA)
    {   // values/keys: M=32768 N=256 K=512 -> waves = (M/16)*(N/64)
        unsigned waves = (SB_ / 16) * (256 / 64);
        gemm_bf16_wmma<<<cdiv(waves, 8), 256, 0, stream>>>(Xbf, 512, WVALb, 512, VAL, 256, b_val, 256, SB_, 256, 512);
        gemm_bf16_wmma<<<cdiv(waves, 8), 256, 0, stream>>>(Xbf, 512, WKEYb, 512, KEY, 256, b_key, 256, SB_, 256, 512);
    }
    rownorm64<<<cdiv((size_t)SB_ * NT_, 256), 256, 0, stream>>>(KEY, SB_ * NT_);
    {   // gx: M=32768 N=2048 K=512 (bias = b_ih + b_hh)
        unsigned waves = (SB_ / 16) * (2048 / 64);
        gemm_bf16_wmma<<<cdiv(waves, 8), 256, 0, stream>>>(Xbf, 512, WIHb, 512, GX, 2048, BIASG, 2048, SB_, 2048, 512);
    }

    // 3) LSTM scan: 1024 stream-ordered steps (64 waves each)
    {
        const __bf16* hp = HZERO;
        for (int s = 0; s < S_; ++s) {
            lstm_step<<<8, 256, 0, stream>>>(GX + (size_t)s * B_ * 2048, hp, WHHb, CST,
                                             Hbf + (size_t)s * B_ * 512);
            hp = Hbf + (size_t)s * B_ * 512;
        }
    }

    // 4) hidden projections (WMMA)
    {   // actions raw: N=64 (24 real, padded)
        unsigned waves = (SB_ / 16) * (64 / 64);
        gemm_bf16_wmma<<<cdiv(waves, 8), 256, 0, stream>>>(Hbf, 512, WACTb, 512, RACT, 64, b_act, 24, SB_, 64, 512);
    }
    {   // queries: N=256
        unsigned waves = (SB_ / 16) * (256 / 64);
        gemm_bf16_wmma<<<cdiv(waves, 8), 256, 0, stream>>>(Hbf, 512, WQb, 512, QRY, 256, b_q, 256, SB_, 256, 512);
    }
    rownorm64<<<cdiv((size_t)SB_ * NT_, 256), 256, 0, stream>>>(QRY, SB_ * NT_);
    postact<<<cdiv((size_t)SB_ * NT_, 256), 256, 0, stream>>>(RACT, ACTS, SB_ * NT_);

    // 5) tape scan: 128 independent (b,nt) units; writes read_outs + final tape
    float* out_main = (float*)d_out;                       // (S,B,512)
    float* out_tape = out_main + (size_t)SB_ * D_;         // (64,B,256)
    turing_scan<<<B_ * NT_, 64, 0, stream>>>(VAL, KEY, QRY, ACTS, ROUT, out_tape);

    // 6) output projection (WMMA): (32768 x 256) @ (256 x 512) + b_out -> d_out
    cvt_f32_bf16<<<cdiv((size_t)SB_ * 256, 256), 256, 0, stream>>>(ROUT, ROUTB, (size_t)SB_ * 256);
    {
        unsigned waves = (SB_ / 16) * (512 / 64);
        gemm_bf16_wmma<<<cdiv(waves, 8), 256, 0, stream>>>(ROUTB, 256, WOUTb, 256, out_main, 512, b_out, 512, SB_, 512, 256);
    }
}
